// NeuralNetwork_19344532701717
// MI455X (gfx1250) — matrix operand order repaired and checked by reading, not compiled, not run on hardware
//
#include <hip/hip_runtime.h>

#define VOCAB 10000
#define DD 32
#define BB 64
#define TT 1024
#define QUEUE_LEN 100

typedef __attribute__((ext_vector_type(16))) __bf16 v16bf;
typedef __attribute__((ext_vector_type(8)))  __bf16 v8bf;
typedef __attribute__((ext_vector_type(8)))  float  v8f;

#if __has_builtin(__builtin_amdgcn_exp2f)
#define EXP2F(x) __builtin_amdgcn_exp2f(x)
#else
#define EXP2F(x) exp2f(x)
#endif
#if __has_builtin(__builtin_amdgcn_rcpf)
#define RCPF(x) __builtin_amdgcn_rcpf(x)
#else
#define RCPF(x) (1.0f / (x))
#endif
#define L2E 1.44269504088896341f

static __device__ __forceinline__ float sig_fast(float x) {
  return RCPF(1.f + EXP2F(-x * L2E));           // sigmoid
}
static __device__ __forceinline__ float tanh_fast(float x) {
  return fmaf(2.f, RCPF(1.f + EXP2F(-2.f * x * L2E)), -1.f);  // 2*sig(2x)-1
}

// ---------------------------------------------------------------------------
// WMMA bf16 fragment loaders (CDNA5 16x16x32 layouts, ISA 7.12.2)
// A (16x32 row-major, stride ld elems):
//   lane L: m=L%16, koff=(L/16)*8; VGPR0-3=K[koff..+7], VGPR4-7=K[16+koff..]
// B (32x16, operand stored K-contiguous per output column, stride ld elems):
//   lane L: n=L%16, koff=(L/16)*8; VGPR0-3=K[koff..+7], VGPR4-7=K[16+koff..]
// ---------------------------------------------------------------------------
static __device__ __forceinline__ v16bf load_fragA(const __bf16* base, int ld) {
  int lane = threadIdx.x & 31;
  int m    = lane & 15;
  int koff = (lane >> 4) << 3;
  const __bf16* p = base + m * ld + koff;
  v8bf lo = *(const v8bf*)(p);
  v8bf hi = *(const v8bf*)(p + 16);
  v16bf r;
#pragma unroll
  for (int i = 0; i < 8; i++) { r[i] = lo[i]; r[i + 8] = hi[i]; }
  return r;
}

static __device__ __forceinline__ v16bf load_fragB(const __bf16* base, int ld) {
  int lane  = threadIdx.x & 31;
  int n     = lane & 15;
  int kbase = (lane >> 4) << 3;
  const __bf16* p = base + n * ld + kbase;
  v8bf lo = *(const v8bf*)(p);
  v8bf hi = *(const v8bf*)(p + 16);
  v16bf r;
#pragma unroll
  for (int i = 0; i < 8; i++) { r[i] = lo[i]; r[i + 8] = hi[i]; }
  return r;
}

// ---------------------------------------------------------------------------
// 1) Embedding gather + input projection: xg[row,j] = b[j] + emb[id] . k[:,j]
// ---------------------------------------------------------------------------
__global__ __launch_bounds__(128) void embed_proj_kernel(
    const int* __restrict__ ids, const float* __restrict__ emb,
    const float* __restrict__ k_lstm, const float* __restrict__ b_lstm,
    float* __restrict__ xg) {
  __shared__ __align__(16) float x[DD];
  int row = blockIdx.x;
  int j   = threadIdx.x;
  int id  = ids[row];
  id = (id < VOCAB) ? id : 0;
  if (j < DD) x[j] = emb[(size_t)id * DD + j];
  __syncthreads();
  float acc = b_lstm[j];
#pragma unroll
  for (int k = 0; k < DD; k++) acc += x[k] * k_lstm[k * 128 + j];
  xg[(size_t)row * 128 + j] = acc;
}

// ---------------------------------------------------------------------------
// 2) LSTM scan: ONE WAVE per batch element (single-wave workgroup => barriers
//    are hardware S_NOPs). Thread j keeps all four of its recurrent weight
//    columns rk[:, j+32g] in 128 VGPRs and computes its own 4 gate
//    pre-activations directly -- no z exchange, no multi-wave barriers.
//    Only inter-lane traffic: broadcasting the 32 new h values through LDS
//    (same-wave DS ops are in-order). xg is manually double-buffered so the
//    streaming global loads overlap the 128-FMA chain.
//    Emits h row-major bf16, h transposed bf16, and a log2e-prescaled copy
//    of h (attention Q operand) so softmax exp needs no scaling multiply.
// ---------------------------------------------------------------------------
__global__ __launch_bounds__(32) void lstm_kernel(
    const float* __restrict__ xg, const float* __restrict__ rk,
    __bf16* __restrict__ h16, __bf16* __restrict__ hT16,
    __bf16* __restrict__ hQ16) {
  __shared__ __align__(16) float h_lds[DD];
  int b = blockIdx.x;
  int j = threadIdx.x;
  float rkreg[4][DD];   // rkreg[g][k] = rk[k][j + 32*g]
#pragma unroll
  for (int g = 0; g < 4; g++)
#pragma unroll
    for (int k = 0; k < DD; k++) rkreg[g][k] = rk[k * 128 + g * DD + j];
  float c = 0.f;
  h_lds[j] = 0.f;
  __syncthreads();
  const float* xgb = xg + (size_t)b * TT * 128;
  // prefetch step 0 inputs
  float xi = xgb[j], xf = xgb[DD + j], xc = xgb[2 * DD + j], xo = xgb[3 * DD + j];
  for (int t = 0; t < TT; t++) {
    float zi = xi, zf = xf, zg = xc, zo = xo;
    if (t + 1 < TT) {                       // issue next step's loads early
      const float* xn = xgb + (t + 1) * 128;
      xi = xn[j]; xf = xn[DD + j]; xc = xn[2 * DD + j]; xo = xn[3 * DD + j];
    }
#pragma unroll
    for (int k4 = 0; k4 < 8; k4++) {
      float4 h4 = *(const float4*)&h_lds[k4 * 4];
      zi += h4.x * rkreg[0][4 * k4] + h4.y * rkreg[0][4 * k4 + 1] +
            h4.z * rkreg[0][4 * k4 + 2] + h4.w * rkreg[0][4 * k4 + 3];
      zf += h4.x * rkreg[1][4 * k4] + h4.y * rkreg[1][4 * k4 + 1] +
            h4.z * rkreg[1][4 * k4 + 2] + h4.w * rkreg[1][4 * k4 + 3];
      zg += h4.x * rkreg[2][4 * k4] + h4.y * rkreg[2][4 * k4 + 1] +
            h4.z * rkreg[2][4 * k4 + 2] + h4.w * rkreg[2][4 * k4 + 3];
      zo += h4.x * rkreg[3][4 * k4] + h4.y * rkreg[3][4 * k4 + 1] +
            h4.z * rkreg[3][4 * k4 + 2] + h4.w * rkreg[3][4 * k4 + 3];
    }
    c = sig_fast(zf) * c + sig_fast(zi) * tanh_fast(zg);
    float h = sig_fast(zo) * tanh_fast(c);
    __syncthreads();                        // S_NOP for 1-wave WG; orders LDS
    h_lds[j] = h;
    __syncthreads();
    h16[((size_t)b * TT + t) * DD + j]  = (__bf16)h;
    hT16[((size_t)b * DD + j) * TT + t] = (__bf16)h;
    hQ16[((size_t)b * TT + t) * DD + j] = (__bf16)(h * L2E);
  }
}

// ---------------------------------------------------------------------------
// 3) Flash attention, 1 wave per 16-query tile, key tiles of 32.
//    |h|<1 => |score|<32 => exp2(score*log2e) <= 2^47: fits f32 AND bf16, so
//    softmax needs NO max subtraction and NO per-iteration rescaling. Q is
//    pre-scaled by log2e so scores land directly in the exp2 domain.
//    Row denominators accumulate as per-lane partials (C-layout) and reduce
//    across the 16-lane group ONCE after the key loop. bk0/bk1 use an
//    even/odd key interleave so each lane's two probabilities are adjacent
//    staged columns -> packed b32 LDS stores.
// ---------------------------------------------------------------------------
__global__ __launch_bounds__(128) void attn_kernel(
    const __bf16* __restrict__ h16, const __bf16* __restrict__ hT16,
    const __bf16* __restrict__ hQ16, __bf16* __restrict__ ctxT16) {
  __shared__ __align__(16) __bf16 plds[4][16 * 40];  // stride 40: 16B-aligned rows
  int w    = threadIdx.x >> 5;
  int lane = threadIdx.x & 31;
  int gw   = blockIdx.x * 4 + w;
  int b    = gw >> 6;
  int q0   = (gw & 63) << 4;
  const __bf16* hb  = h16 + (size_t)b * TT * DD;
  const __bf16* hTb = hT16 + (size_t)b * DD * TT;
  __bf16* pl = &plds[w][0];

  v16bf aq = load_fragA(hQ16 + ((size_t)b * TT + q0) * DD, DD);  // Q * log2e
  v8f acc0 = {}, acc1 = {}, lsum = {};
  int mbase = (lane >> 4) << 3;
  int col   = lane & 15;

  for (int c0 = 0; c0 < TT; c0 += 32) {
    // even keys (c0+2n) and odd keys (c0+2n+1) as S-tile columns
    v16bf bk0 = load_fragB(hb + c0 * DD, 2 * DD);
    v16bf bk1 = load_fragB(hb + (c0 + 1) * DD, 2 * DD);
    v8f zc = {};
    v8f s0 = __builtin_amdgcn_wmma_f32_16x16x32_bf16(false, aq, false, bk0,
                                                     (short)0, zc, false, false);
    v8f s1 = __builtin_amdgcn_wmma_f32_16x16x32_bf16(false, aq, false, bk1,
                                                     (short)0, zc, false, false);
#pragma unroll
    for (int r = 0; r < 8; r++) {
      float e0 = EXP2F(s0[r]);             // scores already in log2 domain
      float e1 = EXP2F(s1[r]);
      lsum[r] += e0 + e1;
      union { __bf16 h2[2]; unsigned u; } pk;
      pk.h2[0] = (__bf16)e0;               // staged col 2*col   (key c0+2col)
      pk.h2[1] = (__bf16)e1;               // staged col 2*col+1 (key c0+2col+1)
      *(unsigned*)&pl[(mbase + r) * 40 + 2 * col] = pk.u;
    }
    v16bf ap  = load_fragA(pl, 40);                 // C->A relayout via LDS
    v16bf bv0 = load_fragB(hTb + c0, TT);           // features 0..15
    v16bf bv1 = load_fragB(hTb + 16 * TT + c0, TT); // features 16..31
    acc0 = __builtin_amdgcn_wmma_f32_16x16x32_bf16(false, ap, false, bv0,
                                                   (short)0, acc0, false, false);
    acc1 = __builtin_amdgcn_wmma_f32_16x16x32_bf16(false, ap, false, bv1,
                                                   (short)0, acc1, false, false);
  }
  // one deferred 16-lane reduction of the softmax denominators
#pragma unroll
  for (int r = 0; r < 8; r++) {
    float s = lsum[r];
#pragma unroll
    for (int msk = 1; msk < 16; msk <<= 1) s += __shfl_xor(s, msk, 32);
    float inv = RCPF(s);
    int m = q0 + mbase + r;
    ctxT16[((size_t)b * DD + col) * TT + m]      = (__bf16)(acc0[r] * inv);
    ctxT16[((size_t)b * DD + col + 16) * TT + m] = (__bf16)(acc1[r] * inv);
  }
}

// ---------------------------------------------------------------------------
// 4) Head: y[b] = ctx^T @ h (32x32, K=1024) via 4 WMMA C-tiles, then
//    out = sigmoid(relu(y@W1+b1)@W2+b2) in VALU. One wave per batch.
// ---------------------------------------------------------------------------
__global__ __launch_bounds__(32) void head_kernel(
    const __bf16* __restrict__ ctxT16, const __bf16* __restrict__ hT16,
    const float* __restrict__ W1, const float* __restrict__ b1,
    const float* __restrict__ W2, const float* __restrict__ b2,
    float* __restrict__ out) {
  __shared__ __align__(16) float y_lds[DD][DD + 1];
  int bb   = blockIdx.x;
  int lane = threadIdx.x;
  const __bf16* cb  = ctxT16 + (size_t)bb * DD * TT;
  const __bf16* hTb = hT16 + (size_t)bb * DD * TT;
  v8f c00 = {}, c01 = {}, c10 = {}, c11 = {};
  for (int t0 = 0; t0 < TT; t0 += 32) {
    v16bf a0  = load_fragA(cb + t0, TT);            // d rows 0..15
    v16bf a1  = load_fragA(cb + 16 * TT + t0, TT);  // d rows 16..31
    v16bf fb0 = load_fragB(hTb + t0, TT);           // e cols 0..15
    v16bf fb1 = load_fragB(hTb + 16 * TT + t0, TT); // e cols 16..31
    c00 = __builtin_amdgcn_wmma_f32_16x16x32_bf16(false, a0, false, fb0, (short)0, c00, false, false);
    c01 = __builtin_amdgcn_wmma_f32_16x16x32_bf16(false, a0, false, fb1, (short)0, c01, false, false);
    c10 = __builtin_amdgcn_wmma_f32_16x16x32_bf16(false, a1, false, fb0, (short)0, c10, false, false);
    c11 = __builtin_amdgcn_wmma_f32_16x16x32_bf16(false, a1, false, fb1, (short)0, c11, false, false);
  }
  int mb  = (lane >> 4) << 3;
  int col = lane & 15;
#pragma unroll
  for (int r = 0; r < 8; r++) {
    y_lds[mb + r][col]           = c00[r];
    y_lds[mb + r][col + 16]      = c01[r];
    y_lds[16 + mb + r][col]      = c10[r];
    y_lds[16 + mb + r][col + 16] = c11[r];
  }
  __syncthreads();
  float w2  = W2[lane];
  float bv1 = b1[lane];
  float bv2 = b2[0];
  for (int d = 0; d < DD; d++) {
    float acc = bv1;
#pragma unroll
    for (int e = 0; e < DD; e++) acc += y_lds[d][e] * W1[e * DD + lane];
    float tsum = fmaxf(acc, 0.f) * w2;
#pragma unroll
    for (int off = 16; off; off >>= 1) tsum += __shfl_xor(tsum, off, 32);
    if (lane == 0) out[bb * DD + d] = sig_fast(tsum + bv2);
  }
}

// ---------------------------------------------------------------------------
// 5) Zero-index queue: single-WG ordered scan, mod-100 ring buffer.
// ---------------------------------------------------------------------------
__global__ __launch_bounds__(1024) void queue_kernel(const int* __restrict__ ids,
                                                     int* __restrict__ outq) {
  __shared__ int sc[1024];
  __shared__ int ring[QUEUE_LEN];
  __shared__ int base;
  int tid = threadIdx.x;
  if (tid == 0) base = 0;
  __syncthreads();
  for (int chunk = 0; chunk < (BB * TT) / 1024; chunk++) {
    int idx = chunk * 1024 + tid;
    int tok = ids[idx];
    tok = (tok < VOCAB) ? tok : 0;
    int flag = (tok == 0) ? 1 : 0;
    sc[tid] = flag;
    __syncthreads();
    for (int off = 1; off < 1024; off <<= 1) {
      int v = sc[tid];
      if (tid >= off) v += sc[tid - off];
      __syncthreads();
      sc[tid] = v;
      __syncthreads();
    }
    int chunk_total = sc[1023];
    if (flag) {
      int rank = base + sc[tid] - 1;
      if (rank >= base + chunk_total - QUEUE_LEN) ring[rank % QUEUE_LEN] = idx;
    }
    __syncthreads();
    if (tid == 0) base += chunk_total;
    __syncthreads();
  }
  int Z = base;
  if (tid < QUEUE_LEN) {
    int r = Z - QUEUE_LEN + tid;
    int i = -1, jj = -1;
    if (r >= 0) {
      int pos = ring[r % QUEUE_LEN];
      i  = pos >> 10;   // T == 1024
      jj = pos & 1023;
    }
    outq[tid * 2]     = i;
    outq[tid * 2 + 1] = jj;
  }
}

// ---------------------------------------------------------------------------
extern "C" void kernel_launch(void* const* d_in, const int* in_sizes, int n_in,
                              void* d_out, int out_size, void* d_ws, size_t ws_size,
                              hipStream_t stream) {
  const int*   ids     = (const int*)d_in[0];
  const float* emb     = (const float*)d_in[1];
  const float* k_lstm  = (const float*)d_in[2];
  const float* rk_lstm = (const float*)d_in[3];
  const float* b_lstm  = (const float*)d_in[4];
  const float* W1      = (const float*)d_in[5];
  const float* b1      = (const float*)d_in[6];
  const float* W2      = (const float*)d_in[7];
  const float* b2      = (const float*)d_in[8];
  float* out = (float*)d_out;

  char* ws = (char*)d_ws;
  float*  xg     = (float*)ws;                          // 64*1024*128*4 = 32 MiB
  __bf16* h16    = (__bf16*)(ws + (size_t)33554432);    // 4 MiB
  __bf16* hT16   = (__bf16*)(ws + (size_t)37748736);    // 4 MiB
  __bf16* ctxT16 = (__bf16*)(ws + (size_t)41943040);    // 4 MiB
  __bf16* hQ16   = (__bf16*)(ws + (size_t)46137344);    // 4 MiB

  embed_proj_kernel<<<BB * TT, 128, 0, stream>>>(ids, emb, k_lstm, b_lstm, xg);
  lstm_kernel<<<BB, 32, 0, stream>>>(xg, rk_lstm, h16, hT16, hQ16);
  attn_kernel<<<(BB * TT / 16) / 4, 128, 0, stream>>>(h16, hT16, hQ16, ctxT16);
  head_kernel<<<BB, 32, 0, stream>>>(ctxT16, hT16, W1, b1, W2, b2, out);
  queue_kernel<<<1, 1024, 0, stream>>>(ids, (int*)(out + BB * DD));
}
